// VectorQuantizer_28698971472261
// MI455X (gfx1250) — matrix operand order repaired and checked by reading, not compiled, not run on hardware
//
#include <hip/hip_runtime.h>
#include <stdint.h>

#define D_ROWS 32768
#define K_ROWS 8192
#define L_COLS 512

#define KT      32          // codebook columns per LDS tile
#define ROW_U4  65          // padded row stride (64 data uint4 + 1 pad) -> 260 dwords, bank-conflict free

typedef _Float16 v16h __attribute__((ext_vector_type(16)));
typedef float    v8f  __attribute__((ext_vector_type(8)));
typedef int      v4i  __attribute__((ext_vector_type(4)));

#define AS1 __attribute__((address_space(1)))
#define AS3 __attribute__((address_space(3)))

// f32 -> f16 bits (RNE via hardware convert)
__device__ __forceinline__ uint16_t f2h(float f) {
  union { _Float16 h; uint16_t u; } v;
  v.h = (_Float16)f;
  return v.u;
}

// issue one per-lane 16B async DMA: global -> LDS (ASYNCcnt tracked)
__device__ __forceinline__ void async_copy16(const uint4* gptr, const uint4* lptr) {
  uint32_t loff = (uint32_t)(uintptr_t)lptr;   // low 32 bits of generic = LDS offset
#if __has_builtin(__builtin_amdgcn_global_load_async_to_lds_b128)
  __builtin_amdgcn_global_load_async_to_lds_b128(
      (AS1 v4i*)(uintptr_t)gptr, (AS3 v4i*)loff, 0, 0);
#else
  asm volatile("global_load_async_to_lds_b128 %0, %1, off"
               :: "v"(loff), "v"((const void*)gptr) : "memory");
#endif
}

__device__ __forceinline__ void wait_async0() {
#if __has_builtin(__builtin_amdgcn_s_wait_asynccnt)
  __builtin_amdgcn_s_wait_asynccnt(0);
#else
  asm volatile("s_wait_asynccnt 0x0" ::: "memory");
#endif
}

// One wave per row of 512 f32: L2-normalize, write f32 (optional) + f16 packed.
__global__ void __launch_bounds__(256)
vq_normalize_kernel(const float* __restrict__ src,
                    float* __restrict__ dst_f32,
                    uint16_t* __restrict__ dst_f16,
                    int nrows) {
  int row  = blockIdx.x * 8 + (threadIdx.x >> 5);
  int lane = threadIdx.x & 31;
  if (row >= nrows) return;

  const float4* s4 = (const float4*)(src + (size_t)row * L_COLS);
  float4 v[4];
  float ss = 0.f;
#pragma unroll
  for (int j = 0; j < 4; ++j) {
    v[j] = s4[lane + 32 * j];
    ss += v[j].x * v[j].x + v[j].y * v[j].y + v[j].z * v[j].z + v[j].w * v[j].w;
  }
#pragma unroll
  for (int off = 16; off >= 1; off >>= 1)
    ss += __shfl_xor(ss, off, 32);

  float scale = 1.0f / (sqrtf(ss) + 1e-8f);

  float4*  df = dst_f32 ? (float4*)(dst_f32 + (size_t)row * L_COLS) : nullptr;
  ushort4* dh = (ushort4*)(dst_f16 + (size_t)row * L_COLS);
#pragma unroll
  for (int j = 0; j < 4; ++j) {
    float4 o;
    o.x = v[j].x * scale; o.y = v[j].y * scale;
    o.z = v[j].z * scale; o.w = v[j].w * scale;
    if (df) df[lane + 32 * j] = o;
    ushort4 b;
    b.x = f2h(o.x); b.y = f2h(o.y); b.z = f2h(o.z); b.w = f2h(o.w);
    dh[lane + 32 * j] = b;
  }
}

// Fused f16 WMMA GEMM + row-argmax over K.
// 8 waves x 16 rows; A fragments resident in VGPRs for the whole K sweep.
// Codebook tiles (32 rows x 512 f16) stream global->LDS via async DMA,
// double-buffered, one barrier per tile.
__global__ void __launch_bounds__(256)
vq_argmax_kernel(const uint16_t* __restrict__ xn,
                 const uint16_t* __restrict__ cbn,
                 int* __restrict__ bestIdx) {
  __shared__ uint4 ldsB[2][KT * ROW_U4];

  const int tid  = threadIdx.x;
  const int lane = tid & 31;
  const int wave = tid >> 5;
  const int rowBase = blockIdx.x * 128 + wave * 16;

  union FragU { uint4 q[2]; v16h v; };

  // ---- preload all 16 A fragments for this wave's 16 rows (ISA A-layout) ----
  const uint4* xn4 = (const uint4*)xn;               // 8 f16 per uint4
  const int arow = rowBase + (lane & 15);
  const int aSel = (lane & 16) ? 8 : 0;              // lanes 16-31: K+8 / K+24
  FragU afrag[16];
#pragma unroll
  for (int i = 0; i < 16; ++i) {
    int e0 = i * 32 + aSel;
    size_t base = (size_t)arow * 64 + (e0 >> 3);
    afrag[i].q[0] = xn4[base];                       // K = e0   .. e0+7
    afrag[i].q[1] = xn4[base + 2];                   // K = e0+16.. e0+23
  }

  float best[8];
  int   bidx[8];
#pragma unroll
  for (int r = 0; r < 8; ++r) { best[r] = -1e30f; bidx[r] = 0; }

  // async tile copy mapping: KT rows x 64 uint4 ; thread -> row = tid>>3, 8 chunks
  const uint4* cbn4 = (const uint4*)cbn;
  const int crow = tid >> 3;                         // 0..31
  const int cj   = tid & 7;                          // 0..7

  // ISA B-layout: lane<16 -> col n, K=0..7,16..23 ; lane>=16 -> col n-16, K=8..15,24..31
  const int brow  = lane & 15;
  const int bSel  = (lane & 16) ? 1 : 0;             // uint4 offset (8 f16)
  const int myCol = lane & 15;

  const int T = K_ROWS / KT;

  // prologue: DMA tile 0 into buffer 0
  {
    const uint4* g = cbn4 + (size_t)crow * 64 + cj;
    const uint4* l = &ldsB[0][crow * ROW_U4 + cj];
#pragma unroll
    for (int m = 0; m < 8; ++m) async_copy16(g + 8 * m, l + 8 * m);
  }

  for (int t = 0; t < T; ++t) {
    wait_async0();                                   // my tile-t DMA landed
    __syncthreads();                                 // everyone's landed; old reads done

    if (t + 1 < T) {                                 // DMA next tile into other buffer
      const uint4* g = cbn4 + (size_t)((t + 1) * KT + crow) * 64 + cj;
      const uint4* l = &ldsB[(t + 1) & 1][crow * ROW_U4 + cj];
#pragma unroll
      for (int m = 0; m < 8; ++m) async_copy16(g + 8 * m, l + 8 * m);
    }

    const uint4* buf = ldsB[t & 1];
#pragma unroll
    for (int ks = 0; ks < KT; ks += 16) {
      v8f c = {0.f, 0.f, 0.f, 0.f, 0.f, 0.f, 0.f, 0.f};
#pragma unroll
      for (int i = 0; i < 16; ++i) {                 // sweep L=512 in K=32 steps
        FragU b;
        const uint4* p = &buf[(ks + brow) * ROW_U4 + i * 4 + bSel];
        b.q[0] = p[0];
        b.q[1] = p[2];
        c = __builtin_amdgcn_wmma_f32_16x16x32_f16(
                false, afrag[i].v, false, b.v, (short)0, c, false, false);
      }
      // running argmax (strict > keeps earliest index, matching argmin-first)
#pragma unroll
      for (int r = 0; r < 8; ++r) {
        if (c[r] > best[r]) { best[r] = c[r]; bidx[r] = t * KT + ks + myCol; }
      }
    }
  }

  // C layout: VGPR r / lanes 0-15 -> row r ; lanes 16-31 -> row r+8.
#pragma unroll
  for (int r = 0; r < 8; ++r) {
    float v  = best[r];
    int   id = bidx[r];
#pragma unroll
    for (int off = 8; off >= 1; off >>= 1) {
      float ov  = __shfl_xor(v, off, 16);
      int   oid = __shfl_xor(id, off, 16);
      if (ov > v || (ov == v && oid < id)) { v = ov; id = oid; }
    }
    if ((lane & 15) == 0) {
      int row = rowBase + r + ((lane & 16) ? 8 : 0);
      bestIdx[row] = id;
    }
  }
}

// z_q = z = codebook[idx] (unnormalized); also write indices as floats.
__global__ void __launch_bounds__(256)
vq_gather_kernel(const float* __restrict__ codebook,
                 const int* __restrict__ bestIdx,
                 float* __restrict__ out) {
  size_t gid = (size_t)blockIdx.x * blockDim.x + threadIdx.x;  // over D*L/4
  size_t e = gid * 4;
  int d = (int)(e >> 9);
  int c = (int)((e & 511) >> 2);
  int idx = bestIdx[d];
  const float4* cb4 = (const float4*)codebook;
  float4 val = cb4[(size_t)idx * 128 + c];
  float4* o0 = (float4*)out;
  float4* o1 = (float4*)(out + (size_t)D_ROWS * L_COLS);
  o0[gid] = val;   // z_q_DL
  o1[gid] = val;   // z_DL
  if (gid < D_ROWS)
    out[(size_t)3 * D_ROWS * L_COLS + gid] = (float)bestIdx[gid];
}

extern "C" void kernel_launch(void* const* d_in, const int* in_sizes, int n_in,
                              void* d_out, int out_size, void* d_ws, size_t ws_size,
                              hipStream_t stream) {
  const float* x  = (const float*)d_in[0];
  const float* cb = (const float*)d_in[1];
  float* out = (float*)d_out;

  char* ws = (char*)d_ws;
  uint16_t* xn   = (uint16_t*)ws;                                   // D*512 f16
  uint16_t* cbn  = (uint16_t*)(ws + (size_t)D_ROWS * L_COLS * 2);   // K*512 f16
  int*      bidx = (int*)(ws + (size_t)(D_ROWS + K_ROWS) * L_COLS * 2);

  float* x_out = out + (size_t)2 * D_ROWS * L_COLS;                 // x (output 3)

  vq_normalize_kernel<<<D_ROWS / 8, 256, 0, stream>>>(x, x_out, xn, D_ROWS);
  vq_normalize_kernel<<<K_ROWS / 8, 256, 0, stream>>>(cb, nullptr, cbn, K_ROWS);
  vq_argmax_kernel<<<D_ROWS / 128, 256, 0, stream>>>(xn, cbn, bidx);
  vq_gather_kernel<<<(D_ROWS * L_COLS / 4) / 256, 256, 0, stream>>>(cb, bidx, out);
}
